// MaskedConv2d_30846455120203
// MI455X (gfx1250) — matrix operand order repaired and checked by reading, not compiled, not run on hardware
//
#include <hip/hip_runtime.h>
#include <hip/hip_bf16.h>

// ---------------- problem constants (fixed by reference) ----------------
#define C_IN   256
#define HW     56
#define IMG    (HW * HW)          // 3136
#define N_IMG  16
#define OC     256
#define KDIM   1152
#define M_TOT  (N_IMG * IMG)      // 50176

// ---------------- tiling ----------------
#define BM 128
#define BN 256                    // full OC: every x element gathered once grid-wide
#define BK 32
#define KTILES (KDIM / BK)        // 36
#define THREADS 512               // 16 waves: 4 (M) x 4 (N)

#define LDA 40                    // bf16 elems per LDS tile row (32 + 8 pad = 80 B pitch)
#define LDC 129                   // f32 elems per LDS C row (128 + 1 pad)

typedef __attribute__((ext_vector_type(16))) __bf16 v16bf;
typedef __attribute__((ext_vector_type(8)))  float  v8f;
typedef __attribute__((ext_vector_type(4)))  unsigned int u32x4;
typedef __attribute__((ext_vector_type(8)))  unsigned int u32x8;
typedef __attribute__((address_space(3)))    unsigned short* lds_as3_t;

union FragBF { uint4 q[2]; v16bf v; };

__device__ __forceinline__ unsigned int pk_bf16(float lo, float hi) {
  __hip_bfloat162 b2 = __float22bfloat162_rn(make_float2(lo, hi));
  unsigned int u;
  __builtin_memcpy(&u, &b2, 4);
  return u;
}

__device__ __forceinline__ unsigned short bf16_1(float v) {
  __hip_bfloat16 b = __float2bfloat16(v);
  unsigned short u;
  __builtin_memcpy(&u, &b, 2);
  return u;
}

// -------- one-time weight convert + transpose: w f32 [K][OC] -> wt bf16 [OC][K] --------
__global__ __launch_bounds__(256)
void wt_transpose(const float* __restrict__ w, unsigned short* __restrict__ wt) {
  __shared__ unsigned short tile[64][72];     // [k][oc], padded
  const int k0 = (int)blockIdx.x * 64;
  const int o0 = (int)blockIdx.y * 64;
  const int c  = (int)threadIdx.x & 63;
  const int r0 = ((int)threadIdx.x >> 6) * 16;

  #pragma unroll
  for (int p = 0; p < 16; ++p) {
    int r = r0 + p;                                      // k in tile
    tile[r][c] = bf16_1(w[(k0 + r) * OC + o0 + c]);      // coalesced along oc
  }
  __syncthreads();
  #pragma unroll
  for (int p = 0; p < 16; ++p) {
    int oc = r0 + p;                                     // oc in tile
    wt[(o0 + oc) * KDIM + k0 + c] = tile[c][oc];         // coalesced along k
  }
}

// -------------------------------- main GEMM kernel --------------------------------
// USE_WT=1: weights pre-converted (bf16, K-major) in ws; W tile staged by the
//           Tensor Data Mover (tensor_load_to_lds, TENSORcnt).
// USE_WT=0: no workspace; W tile converted in-loop by the SIMDs.
template <bool USE_WT>
__global__ __launch_bounds__(THREADS)
void masked_conv_wmma(const float* __restrict__ x,
                      const float* __restrict__ w,               // f32 [K][OC]
                      const unsigned short* __restrict__ wt,     // bf16 [OC][K] (USE_WT)
                      const int*   __restrict__ idx,
                      float*       __restrict__ out)
{
  // phase 1: lds_a (10240) + lds_w (20480) + kmeta (512) = 31232 B
  // phase 2: lds_c 128*129*4 = 66048 B   (aliased union)
  __shared__ __align__(16) unsigned char smem[BM * LDC * 4];  // 66048 B
  __shared__ int4 rowmeta[BM];                                 // never aliased

  unsigned short* lds_a = (unsigned short*)smem;                     // [BM][LDA]
  unsigned short* lds_w = (unsigned short*)(smem + BM * LDA * 2);    // [BN][LDA]
  int4*           kmeta = (int4*)(smem + (BM + BN) * LDA * 2);       // [BK]
  float*          lds_c = (float*)smem;                              // [BM][LDC]

  const int t  = (int)threadIdx.x;
  const int m0 = (int)blockIdx.x * BM;

  // ---- per-output-row metadata (once per block) ----
  if (t < BM) {
    int m   = m0 + t;
    int n   = m / IMG;
    int pos = m - n * IMG;
    int oh  = pos / HW;
    int ow  = pos - oh * HW;
    int xbase = n * (C_IN * IMG) + oh * HW + ow;  // + kdelta = x index
    int obase = n * (OC * IMG) + pos;             // + oc*IMG = out index
    rowmeta[t] = make_int4(xbase, oh, ow, obase);
  }

  const int lane = t & 31;
  const int wid  = t >> 5;          // 0..15
  const int wm   = wid & 3;         // wave M tile -> rows wm*32
  const int wn   = wid >> 2;        // wave N tile -> cols wn*64 (0..3)
  const int l15  = lane & 15;

  // A staging: thread covers row gia, k's gba*8 .. gba*8+7  (one b128 store)
  const int gia = t & 127;
  const int gba = t >> 7;           // 0..3
  // W staging (no-TDM path): thread covers col giw, k's gbw*16 .. +15
  const int giw = t & 255;
  const int gbw = t >> 8;           // 0..1

  // TDM descriptor pieces (uniform)
  const unsigned lds_w_off = (unsigned)(size_t)(lds_as3_t)lds_w;
  // group1 is fully constant for this kernel:
  //  data_size=2B, pad_enable, pad_interval=16DW(64B), pad_amount=4DW(16B)
  //  tensor_dim0=1152, tensor_dim1=256, tile_dim0=32, tile_dim1=256, stride0=1152
  const u32x8 g1 = { 0x06D10000u,            // mask=0 | ds=1<<16 | pad_en | int=3 | amt=3
                     0x04800000u,            // abar=0 | tensor_dim0.lo16 = 1152
                     0x01000000u,            // tensor_dim0.hi16=0 | tensor_dim1.lo16 = 256
                     0x00200000u,            // tensor_dim1.hi16=0 | tile_dim0 = 32
                     0x00000100u,            // tile_dim1 = 256 | tile_dim2 = 0
                     (unsigned)KDIM,         // tensor_dim0_stride.lo32 = 1152
                     0u, 0u };
  const u32x4 gz = { 0u, 0u, 0u, 0u };       // groups 2/3: zero (2D tile)

  v8f acc[2][4];
  #pragma unroll
  for (int a = 0; a < 2; ++a)
    #pragma unroll
    for (int b = 0; b < 4; ++b) {
      v8f z = {0.f, 0.f, 0.f, 0.f, 0.f, 0.f, 0.f, 0.f};
      acc[a][b] = z;
    }

  for (int kt = 0; kt < KTILES; ++kt) {
    const int k0 = kt * BK;

    // ---- decode 32 pruned-row indices ----
    if (t < BK) {
      int rid = idx[k0 + t];             // = c*9 + kh*3 + kw
      int c   = rid / 9;
      int rem = rid - c * 9;
      int kh  = rem / 3;
      int kw  = rem - kh * 3;
      int dh  = kh - 1, dw = kw - 1;
      kmeta[t] = make_int4(c * IMG + dh * HW + dw, dh, dw, 0);
    }
    __syncthreads();   // kmeta ready; all waves past last tile's reads of lds_a/lds_w

    if (USE_WT) {
      // ---- W tile via Tensor Data Mover: one wave issues the whole 256x32 tile ----
      if (wid == 0) {
        unsigned long long ga = (unsigned long long)(size_t)(wt + k0);
        u32x4 g0 = { 1u,                                   // count=1, user descriptor
                     lds_w_off,                            // LDS dest (bytes)
                     (unsigned)ga,                         // global addr [31:0]
                     ((unsigned)(ga >> 32) & 0x1FFFFFFu) | 0x80000000u }; // [56:32] | type=2
        asm volatile("tensor_load_to_lds %0, %1, %2, %3"
                     :: "s"(g0), "s"(g1), "s"(gz), "s"(gz) : "memory");
      }
    } else {
      // ---- stage W tile (BK x BN) K-transposed, converting f32->bf16 in-loop ----
      unsigned int pk[8];
      #pragma unroll
      for (int p = 0; p < 8; ++p) {
        int kk = gbw * 16 + p * 2;
        float v0 = w[(k0 + kk    ) * OC + giw];
        float v1 = w[(k0 + kk + 1) * OC + giw];
        pk[p] = pk_bf16(v0, v1);
      }
      uint4* dst = (uint4*)(lds_w + giw * LDA + gbw * 16);
      dst[0] = make_uint4(pk[0], pk[1], pk[2], pk[3]);
      dst[1] = make_uint4(pk[4], pk[5], pk[6], pk[7]);
    }

    // ---- gather A tile (BM x BK) -> lds_a bf16 (8 elems / thread) ----
    {
      int4 rm = rowmeta[gia];
      unsigned int pk[4];
      #pragma unroll
      for (int p = 0; p < 4; ++p) {
        int e = p * 2;
        int4 k0m = kmeta[gba * 8 + e];
        int4 k1m = kmeta[gba * 8 + e + 1];
        int  ih0 = rm.y + k0m.y, iw0 = rm.z + k0m.z;
        int  ih1 = rm.y + k1m.y, iw1 = rm.z + k1m.z;
        bool ok0 = ((unsigned)ih0 < (unsigned)HW) & ((unsigned)iw0 < (unsigned)HW);
        bool ok1 = ((unsigned)ih1 < (unsigned)HW) & ((unsigned)iw1 < (unsigned)HW);
        float v0 = ok0 ? x[rm.x + k0m.x] : 0.0f;
        float v1 = ok1 ? x[rm.x + k1m.x] : 0.0f;
        pk[p] = pk_bf16(v0, v1);
      }
      *(uint4*)(lds_a + gia * LDA + gba * 8) = make_uint4(pk[0], pk[1], pk[2], pk[3]);
    }

    if (USE_WT) {
      if (wid == 0) __builtin_amdgcn_s_wait_tensorcnt(0);  // W tile landed in LDS
    }
    __syncthreads();   // tiles ready

    // ---- WMMA: 2 (M) x 4 (N) tiles of 16x16x32 ----
    const int ahalf = (lane < 16) ? 0 : 8;    // A: K0-7/16-23 vs K8-15/24-31
    const int bhalf = (lane < 16) ? 0 : 8;    // B: K0-7/16-23 vs K8-15/24-31

    FragBF afr[2];
    #pragma unroll
    for (int tm = 0; tm < 2; ++tm) {
      const unsigned short* src = lds_a + (wm * 32 + tm * 16 + l15) * LDA + ahalf;
      afr[tm].q[0] = *(const uint4*)(src);
      afr[tm].q[1] = *(const uint4*)(src + 16);
    }
    #pragma unroll
    for (int tn = 0; tn < 4; ++tn) {
      FragBF bfr;
      const unsigned short* src = lds_w + (wn * 64 + tn * 16 + l15) * LDA + bhalf;
      bfr.q[0] = *(const uint4*)(src);
      bfr.q[1] = *(const uint4*)(src + 16);
      #pragma unroll
      for (int tm = 0; tm < 2; ++tm) {
        acc[tm][tn] = __builtin_amdgcn_wmma_f32_16x16x32_bf16(
            false, afr[tm].v, false, bfr.v, (short)0, acc[tm][tn], false, false);
      }
    }
    // next iteration's first barrier separates these reads from tile overwrite
  }

  // ---- epilogue: two 128(M) x 128(OC) passes through the LDS C tile ----
  const int i       = t & 127;          // m row (fixed per thread)
  const int colbase = t >> 7;           // 0..3
  const int rbase0  = (lane < 16) ? 0 : 8;

  #pragma unroll
  for (int h = 0; h < 2; ++h) {
    __syncthreads();   // h==0: done with lds_a/lds_w; h==1: pass-0 reads done
    if ((wn >> 1) == h) {
      #pragma unroll
      for (int tm = 0; tm < 2; ++tm)
        #pragma unroll
        for (int tn = 0; tn < 4; ++tn) {
          int col = (wn & 1) * 64 + tn * 16 + l15;
          int row = wm * 32 + tm * 16 + rbase0;
          #pragma unroll
          for (int r = 0; r < 8; ++r)
            lds_c[(row + r) * LDC + col] = acc[tm][tn][r];
        }
    }
    __syncthreads();
    const int obase = rowmeta[i].w;
    #pragma unroll 4
    for (int rep = 0; rep < 32; ++rep) {
      int col = colbase + rep * 4;                   // 0..127 within pass
      int oc  = h * 128 + col;
      __builtin_nontemporal_store(lds_c[i * LDC + col], &out[obase + oc * IMG]);
    }
  }
}

extern "C" void kernel_launch(void* const* d_in, const int* in_sizes, int n_in,
                              void* d_out, int out_size, void* d_ws, size_t ws_size,
                              hipStream_t stream) {
  const float* x   = (const float*)d_in[0];
  const float* w   = (const float*)d_in[1];
  const int*   idx = (const int*)d_in[2];
  float*       out = (float*)d_out;

  dim3 grid(M_TOT / BM, 1, 1);   // 392 blocks
  dim3 block(THREADS, 1, 1);

  const size_t wt_bytes = (size_t)OC * KDIM * 2;   // 589,824 B
  if (ws_size >= wt_bytes) {
    unsigned short* wt = (unsigned short*)d_ws;
    wt_transpose<<<dim3(KDIM / 64, OC / 64, 1), dim3(256, 1, 1), 0, stream>>>(w, wt);
    masked_conv_wmma<true><<<grid, block, 0, stream>>>(x, w, wt, idx, out);
  } else {
    masked_conv_wmma<false><<<grid, block, 0, stream>>>(x, w, nullptr, idx, out);
  }

  (void)in_sizes; (void)n_in; (void)out_size;
}
